// ConvNet_81501299409071
// MI455X (gfx1250) — matrix operand order repaired and checked by reading, not compiled, not run on hardware
//
#include <hip/hip_runtime.h>
#include <cmath>

typedef __attribute__((ext_vector_type(16))) _Float16     v16h;
typedef __attribute__((ext_vector_type(8)))  _Float16     v8h;
typedef __attribute__((ext_vector_type(4)))  _Float16     v4h;
typedef __attribute__((ext_vector_type(8)))  float        v8f;
typedef __attribute__((ext_vector_type(4)))  unsigned int v4u;
typedef __attribute__((ext_vector_type(8)))  int          v8i;
typedef __attribute__((ext_vector_type(4)))  int          v4i;

#define WAVES   4
#define THREADS (WAVES * 32)

#ifndef __has_builtin
#define __has_builtin(x) 0
#endif
#if __has_builtin(__builtin_amdgcn_tensor_load_to_lds) && \
    __has_builtin(__builtin_amdgcn_s_wait_tensorcnt)
#define USE_TDM 1
#else
#define USE_TDM 0
#endif

// ---- workspace layout (units: halves) --------------------------------------
#define OFF_FW1P 0          // fc1 B-fragments: 63 kt * 4 nt * 512 = 129024
#define OFF_W1P  129024     // conv1 B-fragment: 512
#define OFF_W2P  129536     // conv2 B-fragments: 6 * 512 = 3072
#define OFF_FW2P 132608     // fc2 B-fragments: 2 * 512 = 1024

// ---- dynamic LDS layout (units: halves) ------------------------------------
#define L_ACT2   0          // [16 rows][2016]  fc1 A, K-swizzled blocks of 32
#define L_ACT3   32256      // [16 rows][64]    fc2 A, K-swizzled
#define L_WAVE   33280      // per-wave scratch:
#define LW_IMG   0          //   784 halves f16 image
#define LW_C1    800        //   conv1+relu out: 10ch * 576
#define LW_A1    6560       //   pooled: 10ch * 144
#define LW_ZERO  8000       //   zero source (16 halves)
#define LW_TRASH 8016       //   masked-store dump (16 halves)
#define LW_STAGE 8032       //   2 x 784 f32 TDM staging (2*1568 halves)
#define WAVE_LDS 11168
#define LDS_HALVES (L_WAVE + WAVES * WAVE_LDS)   // 77952 halves = 155904 B

static __device__ inline v8f wmma32(v16h a, v16h b, v8f c) {
  return __builtin_amdgcn_wmma_f32_16x16x32_f16(false, a, false, b,
                                                (short)0, c, false, false);
}

// K-swizzle within a 32-block: groups [0,1,2,3] -> positions [0,2,1,3]
// so each lane's 16 A-halves are contiguous (lanes<16: {0..7,16..23}).
static __device__ inline int kswz(int j) {
  const int g = j >> 3;
  return ((((g & 1) << 1) | (g >> 1)) << 3) | (j & 7);
}

#if USE_TDM
// TDM: async DMA of one 784-float image into LDS (D# per ISA 08 §8.3/8.4).
static __device__ inline void tdm_img_load(const float* gsrc, unsigned int ldsByte) {
  const unsigned long long ga = (unsigned long long)(uintptr_t)gsrc;
  v4u g0;
  g0[0] = 1u;                                            // count=1 (valid), user
  g0[1] = ldsByte;                                       // lds_addr
  g0[2] = (unsigned int)(ga & 0xFFFFFFFFu);              // global_addr lo
  g0[3] = (unsigned int)((ga >> 32) & 0x01FFFFFFu)       // global_addr hi
        | 0x80000000u;                                   // type=2 ("image")
  v8i g1;
  g1[0] = (int)(2u << 16);      // workgroup_mask=0, data_size=4B (code 2)
  g1[1] = (int)(784u << 16);    // tensor_dim0[15:0] at bits 63:48
  g1[2] = (int)(1u << 16);      // tensor_dim0 hi=0 | tensor_dim1 lo=1
  g1[3] = (int)(784u << 16);    // tensor_dim1 hi=0 | tile_dim0=784
  g1[4] = 0;                    // tile_dim1/2 unused
  g1[5] = 784;                  // tensor_dim0_stride
  g1[6] = 0;
  g1[7] = 0;
  const v4i z4 = {0, 0, 0, 0};                 // groups 2/3: unused (<=2D)
  const v8i z8 = {0, 0, 0, 0, 0, 0, 0, 0};
  __builtin_amdgcn_tensor_load_to_lds(g0, g1, z4, z4, z8, 0);  // 6-arg form
}
#endif

// ============================================================================
// Prep: E = mean(|w|), binarize + pre-swizzle weights into wave32 B-fragment
// layout: frag[lane][h] = element (k = 32*kt + 8*(lane>=16) + (h&7) + 16*(h>>3),
//                                  n = 16*nt + (lane&15)); pads are zero.
// ============================================================================
__global__ void prep_kernel(const float* __restrict__ w1,
                            const float* __restrict__ w2,
                            const float* __restrict__ fw1,
                            const float* __restrict__ fw2,
                            _Float16* __restrict__ wsH) {
  __shared__ float red[256];
  const int tid = threadIdx.x;

  auto bsum = [&](const float* p, int n) -> float {
    float s = 0.f;
    for (int i = tid; i < n; i += 256) s += fabsf(p[i]);
    red[tid] = s; __syncthreads();
    for (int off = 128; off > 0; off >>= 1) {
      if (tid < off) red[tid] += red[tid + off];
      __syncthreads();
    }
    float r = red[0]; __syncthreads();
    return r;
  };

  const float E1 = bsum(w1, 250)     * (1.f / 250.f);
  const float E2 = bsum(w2, 1800)    * (1.f / 1800.f);
  const float E3 = bsum(fw1, 100000) * (1.f / 100000.f);
  const float E4 = bsum(fw2, 500)    * (1.f / 500.f);

  auto bq = [](float w, float E) -> _Float16 {
    return (_Float16)(w > 0.f ? E : (w < 0.f ? -E : 0.f));
  };

  for (int idx = tid; idx < 512; idx += 256) {            // conv1: K32(25) N16(10)
    const int lane = idx >> 4, h = idx & 15;
    const int k = ((lane >> 4) << 3) + (h & 7) + ((h >> 3) << 4), n = lane & 15;
    _Float16 v = (_Float16)0.f;
    if (k < 25 && n < 10) v = bq(w1[n * 25 + k], E1);
    wsH[OFF_W1P + idx] = v;
  }
  for (int idx = tid; idx < 3072; idx += 256) {           // conv2: K96(90) N32(20)
    const int fi = idx >> 9, r = idx & 511;
    const int lane = r >> 4, h = r & 15;
    const int kt = fi >> 1, nt = fi & 1;
    const int k  = kt * 32 + ((lane >> 4) << 3) + (h & 7) + ((h >> 3) << 4);
    const int co = nt * 16 + (lane & 15);
    _Float16 v = (_Float16)0.f;
    if (k < 90 && co < 20) v = bq(w2[co * 90 + k], E2);
    wsH[OFF_W2P + idx] = v;
  }
  for (int idx = tid; idx < 129024; idx += 256) {         // fc1: K2016(2000) N64(50)
    const int fi = idx >> 9, r = idx & 511;
    const int lane = r >> 4, h = r & 15;
    const int kt = fi >> 2, nt = fi & 3;
    const int k = kt * 32 + ((lane >> 4) << 3) + (h & 7) + ((h >> 3) << 4);
    const int n = nt * 16 + (lane & 15);
    _Float16 v = (_Float16)0.f;
    if (k < 2000 && n < 50) v = bq(fw1[n * 2000 + k], E3);
    wsH[OFF_FW1P + idx] = v;
  }
  for (int idx = tid; idx < 1024; idx += 256) {           // fc2: K64(50) N16(10)
    const int fi = idx >> 9, r = idx & 511;
    const int lane = r >> 4, h = r & 15;
    const int k = fi * 32 + ((lane >> 4) << 3) + (h & 7) + ((h >> 3) << 4);
    const int n = lane & 15;
    _Float16 v = (_Float16)0.f;
    if (k < 50 && n < 10) v = bq(fw2[n * 50 + k], E4);
    wsH[OFF_FW2P + idx] = v;
  }
}

// ============================================================================
// Fused forward: 1 WG (4 wave32) = 16 images. Everything stays in LDS.
// ============================================================================
__global__ __launch_bounds__(THREADS)
void convnet_fused(const float* __restrict__ x,
                   const float* __restrict__ b1,  const float* __restrict__ b2,
                   const float* __restrict__ fb1, const float* __restrict__ fb2,
                   const _Float16* __restrict__ wsH,
                   float* __restrict__ out) {
  extern __shared__ _Float16 sm[];
  const int tid  = threadIdx.x;
  const int lane = tid & 31;
  const int wv   = tid >> 5;
  const int wg   = blockIdx.x;
  const int lhi  = lane >> 4;
  const int lN   = lane & 15;

  _Float16* act2s = sm + L_ACT2;
  _Float16* act3s = sm + L_ACT3;
  _Float16* wbuf  = sm + L_WAVE + wv * WAVE_LDS;
  _Float16* simg  = wbuf + LW_IMG;

  if (lane == 0) wbuf[LW_ZERO] = (_Float16)0.f;

  const float b1v = (lN < 10) ? b1[lN] : 0.f;
  float b2v[2];
  b2v[0] = b2[lN];
  b2v[1] = (16 + lN < 20) ? b2[16 + lN] : 0.f;

  const int img0 = wg * 16 + wv * 4;
#if USE_TDM
  const unsigned int stageByte =
      (unsigned int)((L_WAVE + wv * WAVE_LDS + LW_STAGE) * 2);
  tdm_img_load(x + (size_t)img0 * 784, stageByte);        // prefetch image 0
#endif

  // ======================= conv stage: 4 images per wave ====================
  for (int i = 0; i < 4; ++i) {
    const int li = wv * 4 + i;

#if USE_TDM
    if (i < 3) {            // async-prefetch next image, wait for current
      tdm_img_load(x + (size_t)(img0 + i + 1) * 784,
                   stageByte + (unsigned int)(((i + 1) & 1) * 3136));
      __builtin_amdgcn_s_wait_tensorcnt(1);
    } else {
      __builtin_amdgcn_s_wait_tensorcnt(0);
    }
    const float4* xi4 = (const float4*)(wbuf + LW_STAGE + (i & 1) * 1568);
#else
    const float4* xi4 = (const float4*)(x + (size_t)(img0 + i) * 784);
    if (img0 + i + 1 < (int)gridDim.x * 16)
      __builtin_prefetch(x + (size_t)(img0 + i + 1) * 784 + lane * 32, 0, 0);
#endif
    for (int t = lane; t < 196; t += 32) {                // f32 -> f16
      const float4 v = xi4[t];
      v4h h4; h4[0] = (_Float16)v.x; h4[1] = (_Float16)v.y;
              h4[2] = (_Float16)v.z; h4[3] = (_Float16)v.w;
      *(v4h*)(simg + t * 4) = h4;
    }
    __syncthreads();

    // ---- conv1 as GEMM: M=576, K=32(25), N=16(10); 36 WMMAs ---------------
    {
      const v16h bW1 = *(const v16h*)(wsH + OFF_W1P + lane * 16);
#pragma unroll 4
      for (int mt = 0; mt < 36; ++mt) {
        const int p  = mt * 16 + lN;
        const int oy = p / 24, ox = p % 24;
        v16h a;
#pragma unroll
        for (int h = 0; h < 16; ++h) {
          const int k   = (h & 7) + (lhi << 3) + ((h >> 3) << 4);
          const int ky  = k / 5, kx = k % 5;
          const int idx = (k < 25) ? ((oy + ky) * 28 + ox + kx) : LW_ZERO;
          a[h] = wbuf[idx];                               // branchless gather
        }
        v8f c;
#pragma unroll
        for (int q = 0; q < 8; ++q) c[q] = b1v;
        c = wmma32(a, bW1, c);
        v8h r;
#pragma unroll
        for (int q = 0; q < 8; ++q) r[q] = (_Float16)fmaxf(c[q], 0.f);
        const int sidx = (lN < 10)
            ? (LW_C1 + lN * 576 + mt * 16 + (lhi << 3)) : LW_TRASH;
        *(v8h*)(wbuf + sidx) = r;
      }
    }
    __syncthreads();

    // ---- 2x2 maxpool -> [10][12][12] --------------------------------------
    for (int t = lane; t < 1440; t += 32) {
      const int ch = t / 144, q = t % 144;
      const int py = q / 12, px = q % 12;
      const int bix = LW_C1 + ch * 576 + (py * 2) * 24 + px * 2;
      const float m0 = fmaxf((float)wbuf[bix],      (float)wbuf[bix + 1]);
      const float m1 = fmaxf((float)wbuf[bix + 24], (float)wbuf[bix + 25]);
      wbuf[LW_A1 + ch * 144 + q] = (_Float16)fmaxf(m0, m1);
    }
    __syncthreads();

    // ---- conv2 as GEMM: M=112(100), K=96(90), N=32(20); 42 WMMAs ----------
    {
      v16h bW2[3][2];
#pragma unroll
      for (int kt = 0; kt < 3; ++kt)
#pragma unroll
        for (int nt = 0; nt < 2; ++nt)
          bW2[kt][nt] =
              *(const v16h*)(wsH + OFF_W2P + ((kt * 2 + nt) * 32 + lane) * 16);

      for (int mt = 0; mt < 7; ++mt) {
        const int m  = mt * 16 + lN;
        const int py = m / 10, px = m % 10;
        const bool mok = (m < 100);
        v16h a2[3];
#pragma unroll
        for (int kt = 0; kt < 3; ++kt) {
#pragma unroll
          for (int h = 0; h < 16; ++h) {
            const int k  = kt * 32 + (h & 7) + (lhi << 3) + ((h >> 3) << 4);
            const int ci = k / 9, tp = k % 9;
            const int off = LW_A1 + ci * 144 + (py + tp / 3) * 12 + (px + tp % 3);
            const int idx = (k < 90 && mok) ? off : LW_ZERO;
            a2[kt][h] = wbuf[idx];
          }
        }
#pragma unroll
        for (int nt = 0; nt < 2; ++nt) {
          v8f c;
#pragma unroll
          for (int q = 0; q < 8; ++q) c[q] = b2v[nt];
          c = wmma32(a2[0], bW2[0][nt], c);
          c = wmma32(a2[1], bW2[1][nt], c);
          c = wmma32(a2[2], bW2[2][nt], c);
          const int co = nt * 16 + lN;
          // store K-swizzled so fc1 A-fragments are contiguous v16h loads
#pragma unroll
          for (int g = 0; g < 2; ++g) {
            const int mmb = mt * 16 + (lhi << 3) + g * 4;
            v4h r;
#pragma unroll
            for (int j = 0; j < 4; ++j)
              r[j] = (_Float16)fmaxf(c[g * 4 + j], 0.f);
            const int f  = co * 100 + mmb;                // feature index
            const int sw = ((f >> 5) << 5) + kswz(f & 31);
            _Float16* dst = (mmb < 100 && co < 20)
                ? (act2s + li * 2016 + sw)
                : (wbuf + LW_TRASH);
            *(v4h*)dst = r;
          }
        }
      }
    }
    if (lane < 16)                                        // zero pad k=2000..2015
      act2s[li * 2016 + 62 * 32 + 8 + (lane & 7) + ((lane >> 3) << 4)] =
          (_Float16)0.f;
    __syncthreads();
  }

  // ======================= fc1: [16,2016] x [2016,64] =======================
  const int  n1    = wv * 16 + lN;
  const float fb1v = (n1 < 50) ? fb1[n1] : 0.f;
  v8f acc;
#pragma unroll
  for (int q = 0; q < 8; ++q) acc[q] = fb1v;
  const _Float16* fw1p = wsH + OFF_FW1P;
#pragma unroll 7
  for (int kt = 0; kt < 63; ++kt) {
    // K-swizzled rows -> one contiguous v16h per lane (2x ds_load_b128)
    const v16h a = *(const v16h*)(act2s + lN * 2016 + kt * 32 + (lhi << 4));
    const v16h b = *(const v16h*)(fw1p + ((kt * 4 + wv) * 32 + lane) * 16);
    acc = wmma32(a, b, acc);
  }
  {
    const int kt = n1 >> 5;
    const int sw = (kt << 5) + kswz(n1 & 31);
#pragma unroll
    for (int q = 0; q < 8; ++q)
      act3s[(q + (lhi << 3)) * 64 + sw] = (_Float16)fmaxf(acc[q], 0.f);
  }
  __syncthreads();

  // ======================= fc2 + log_softmax (wave 0) =======================
  if (wv == 0) {
    const v16h bF0 = *(const v16h*)(wsH + OFF_FW2P + lane * 16);
    const v16h bF1 = *(const v16h*)(wsH + OFF_FW2P + (32 + lane) * 16);
    const float fb2v = (lN < 10) ? fb2[lN] : 0.f;
    v8f cl;
#pragma unroll
    for (int q = 0; q < 8; ++q) cl[q] = fb2v;
#pragma unroll
    for (int kt = 0; kt < 2; ++kt) {
      const v16h a = *(const v16h*)(act3s + lN * 64 + kt * 32 + (lhi << 4));
      cl = wmma32(a, kt == 0 ? bF0 : bF1, cl);
    }
    const bool valid = (lN < 10);
#pragma unroll
    for (int q = 0; q < 8; ++q) {
      const float xv = cl[q];
      float mx = valid ? xv : -__builtin_inff();
#pragma unroll
      for (int d = 8; d >= 1; d >>= 1) mx = fmaxf(mx, __shfl_xor(mx, d, 16));
      float s = valid ? __expf(xv - mx) : 0.f;
#pragma unroll
      for (int d = 8; d >= 1; d >>= 1) s += __shfl_xor(s, d, 16);
      if (valid) {
        const int row = q + (lhi << 3);
        out[(size_t)(wg * 16 + row) * 10 + lN] = xv - mx - __logf(s);
      }
    }
  }
}

// ============================================================================
extern "C" void kernel_launch(void* const* d_in, const int* in_sizes, int n_in,
                              void* d_out, int out_size, void* d_ws, size_t ws_size,
                              hipStream_t stream) {
  const float* x   = (const float*)d_in[0];
  const float* w1  = (const float*)d_in[1];
  const float* b1  = (const float*)d_in[2];
  const float* w2  = (const float*)d_in[3];
  const float* b2  = (const float*)d_in[4];
  const float* fw1 = (const float*)d_in[5];
  const float* fb1 = (const float*)d_in[6];
  const float* fw2 = (const float*)d_in[7];
  const float* fb2 = (const float*)d_in[8];
  _Float16* wsH = (_Float16*)d_ws;
  float*    out = (float*)d_out;

  const int B = in_sizes[0] / 784;   // 8192

  prep_kernel<<<1, 256, 0, stream>>>(w1, w2, fw1, fw2, wsH);
  convnet_fused<<<B / 16, THREADS, LDS_HALVES * 2, stream>>>(
      x, b1, b2, fb1, fb2, wsH, out);
}
